// BatchTreeLSTM_5446018531923
// MI455X (gfx1250) — matrix operand order repaired and checked by reading, not compiled, not run on hardware
//
#include <hip/hip_runtime.h>
#include <hip/hip_bf16.h>

// ---------------------------------------------------------------------------
// TreeLSTM on MI455X (gfx1250, wave32, WMMA f32_16x16x32_f16)
//
// Per level d (n = 2^d nodes/tree): GEMM [B*n, 128] @ [128, 64] for gates
// i,o,u (combined = [x | h_l + h_r]) and per-child forget gates
// ([x | h_l] and [x | h_r] @ W_f). One wave = one 16-row node tile,
// 80 v_wmma_f32_16x16x32_f16 per wave (non-leaf). fp32 accumulate, f16 in.
// Activations use the gfx1250 hardware v_tanh_f32 transcendental.
// ---------------------------------------------------------------------------

typedef __attribute__((ext_vector_type(16))) _Float16 v16h;
typedef __attribute__((ext_vector_type(8)))  float    v8f;

#define BATCH 256
#define NODES 4095
#define DEPTH 12
#define HDIM  64

static __device__ __forceinline__ v8f wmma16x16x32(v16h a, v16h b, v8f c) {
    return __builtin_amdgcn_wmma_f32_16x16x32_f16(
        /*neg_a=*/false, a, /*neg_b=*/false, b,
        /*c_mod=*/(short)0, c, /*reuse_a=*/false, /*reuse_b=*/false);
}

// gfx1250 has a hardware TANH transcendental (TRANS32 class).
static __device__ __forceinline__ float fast_tanh(float x) {
#if __has_builtin(__builtin_amdgcn_tanhf)
    return __builtin_amdgcn_tanhf(x);
#else
    return tanhf(x);
#endif
}

// sigmoid(x) = 0.5 * tanh(x/2) + 0.5  -> 1 v_tanh + 1 fma
static __device__ __forceinline__ float fast_sigmoid(float x) {
    return fmaf(0.5f, fast_tanh(0.5f * x), 0.5f);
}

// A-matrix (16x32 f16) per-lane layout (ISA 7.12.2):
//   lane l<16 : M=l,    K in {0..7} U {16..23}
//   lane l>=16: M=l-16, K in {8..15} U {24..31}
// koff = kt*32 + 8*(lane>=16); halves 0..7 <- rowp[koff..koff+7],
// halves 8..15 <- rowp[koff+16..koff+23].
static __device__ __forceinline__ v16h load_a_row(const float* __restrict__ rowp, int koff) {
    const float4* p = reinterpret_cast<const float4*>(rowp + koff);
    const float4* q = reinterpret_cast<const float4*>(rowp + koff + 16);
    float4 q0 = p[0], q1 = p[1], q2 = q[0], q3 = q[1];
    v16h a;
    a[0]=(_Float16)q0.x; a[1]=(_Float16)q0.y; a[2]=(_Float16)q0.z; a[3]=(_Float16)q0.w;
    a[4]=(_Float16)q1.x; a[5]=(_Float16)q1.y; a[6]=(_Float16)q1.z; a[7]=(_Float16)q1.w;
    a[8]=(_Float16)q2.x; a[9]=(_Float16)q2.y; a[10]=(_Float16)q2.z; a[11]=(_Float16)q2.w;
    a[12]=(_Float16)q3.x; a[13]=(_Float16)q3.y; a[14]=(_Float16)q3.z; a[15]=(_Float16)q3.w;
    return a;
}

// Packed f16 weights in B-matrix (32x16) wave layout:
//   lane l<16 : N=l,    K tile-local {0..7} U {16..23}
//   lane l>=16: N=l-16, K tile-local {8..15} U {24..31}
// index((((g*4+kt)*4+nt)*32+lane)*16+j); 32B-aligned per-lane chunk.
static __device__ __forceinline__ v16h load_b_tile(const _Float16* __restrict__ wpack,
                                                   int g, int kt, int nt, int lane) {
    return *reinterpret_cast<const v16h*>(
        wpack + ((size_t)(((g * 4 + kt) * 4 + nt) * 32 + lane)) * 16);
}

__global__ void pack_weights_kernel(const float* __restrict__ Wi,
                                    const float* __restrict__ Wf,
                                    const float* __restrict__ Wo,
                                    const float* __restrict__ Wu,
                                    _Float16* __restrict__ wpack) {
    int t = blockIdx.x * blockDim.x + threadIdx.x;   // 0 .. 32767
    int jj = t & 15;
    int l  = (t >> 4) & 31;
    int nt = (t >> 9) & 3;
    int kt = (t >> 11) & 3;
    int g  = t >> 13;
    int K   = kt * 32 + ((jj < 8) ? jj : jj + 8) + ((l >= 16) ? 8 : 0);
    int col = nt * 16 + (l & 15);
    const float* W = (g == 0) ? Wi : (g == 1) ? Wf : (g == 2) ? Wo : Wu;
    wpack[t] = (_Float16)W[K * HDIM + col];
}

template <bool LEAF>
__global__ __launch_bounds__(128)
void tree_level_kernel(const float* __restrict__ x,
                       const _Float16* __restrict__ wpack,
                       const float* __restrict__ bi, const float* __restrict__ bf,
                       const float* __restrict__ bo, const float* __restrict__ bu,
                       const float* __restrict__ h_in, const float* __restrict__ c_in,
                       float* __restrict__ h_out, float* __restrict__ c_out,
                       int n, int dshift,
                       float* __restrict__ root_out) {
    const int lane = threadIdx.x & 31;
    const int wave = threadIdx.x >> 5;
    const int tile = blockIdx.x * 4 + wave;
    const int row0 = tile * 16;
    const int m    = lane & 15;
    const int hi   = lane >> 4;            // lane half select
    const int koffA = hi * 8;              // A-layout lane K base

    // ---- A-tile construction for this wave's 16 node rows -----------------
    const int row = row0 + m;              // global node row for A loads
    const int b   = row >> dshift;         // tree index  (n = 1<<dshift)
    const int j   = row & (n - 1);         // node within level
    const float* xrow = x + ((size_t)b * NODES + (size_t)(n - 1 + j)) * HDIM;

    v16h ax0 = load_a_row(xrow, 0 + koffA);
    v16h ax1 = load_a_row(xrow, 32 + koffA);
    v16h ahl0, ahl1, ahr0, ahr1, ahs0, ahs1;
    if (!LEAF) {
        const float* hl = h_in + ((size_t)b * (2 * n) + (size_t)(2 * j)) * HDIM;
        const float* hr = hl + HDIM;       // sibling row is adjacent
        ahl0 = load_a_row(hl, 0 + koffA);  ahl1 = load_a_row(hl, 32 + koffA);
        ahr0 = load_a_row(hr, 0 + koffA);  ahr1 = load_a_row(hr, 32 + koffA);
        ahs0 = ahl0 + ahr0;                // h_l + h_r  (packed f16 adds)
        ahs1 = ahl1 + ahr1;
    }

    // ---- 4 output n-tiles of 16 columns each ------------------------------
    #pragma unroll
    for (int nt = 0; nt < 4; ++nt) {
        v8f ai = {}, ao = {}, au = {}, afl = {}, afr = {};
        constexpr int KT = LEAF ? 2 : 4;
        #pragma unroll
        for (int kt = 0; kt < KT; ++kt) {
            v16h a = (kt == 0) ? ax0 : (kt == 1) ? ax1 : (kt == 2) ? ahs0 : ahs1;
            ai = wmma16x16x32(a, load_b_tile(wpack, 0, kt, nt, lane), ai);
            ao = wmma16x16x32(a, load_b_tile(wpack, 2, kt, nt, lane), ao);
            au = wmma16x16x32(a, load_b_tile(wpack, 3, kt, nt, lane), au);
            if (!LEAF) {
                v16h al = (kt == 0) ? ax0 : (kt == 1) ? ax1 : (kt == 2) ? ahl0 : ahl1;
                v16h ar = (kt == 0) ? ax0 : (kt == 1) ? ax1 : (kt == 2) ? ahr0 : ahr1;
                v16h wf = load_b_tile(wpack, 1, kt, nt, lane);
                afl = wmma16x16x32(al, wf, afl);
                afr = wmma16x16x32(ar, wf, afr);
            }
        }

        // Epilogue: C/D layout => VGPR v holds (M = v + 8*hi, N = lane&15).
        const int col = nt * 16 + m;
        const float bbi = bi[col], bbf = bf[col], bbo = bo[col], bbu = bu[col];
        #pragma unroll
        for (int v = 0; v < 8; ++v) {
            const int rowC = row0 + v + hi * 8;
            const int bc = rowC >> dshift;
            const int jc = rowC & (n - 1);
            const float it = fast_sigmoid(ai[v] + bbi);
            const float ot = fast_sigmoid(ao[v] + bbo);
            const float ut = fast_tanh(au[v] + bbu);
            float c;
            if (LEAF) {
                c = it * ut;   // zero children: forget-gate term vanishes
            } else {
                const size_t cb = ((size_t)bc * (2 * n) + (size_t)(2 * jc)) * HDIM + col;
                const float cl = c_in[cb];
                const float cr = c_in[cb + HDIM];
                const float fl = fast_sigmoid(afl[v] + bbf);
                const float fr = fast_sigmoid(afr[v] + bbf);
                c = fmaf(it, ut, fmaf(fl, cl, fr * cr));
            }
            const float h = ot * fast_tanh(c);
            const size_t ob = ((size_t)bc * n + (size_t)jc) * HDIM + col;
            h_out[ob] = h;
            c_out[ob] = c;
            if (root_out) {                // d==0: n==1, jc==0
                root_out[bc * HDIM + col] = h;
                root_out[BATCH * HDIM + bc * HDIM + col] = c;
            }
        }
    }
}

extern "C" void kernel_launch(void* const* d_in, const int* in_sizes, int n_in,
                              void* d_out, int out_size, void* d_ws, size_t ws_size,
                              hipStream_t stream) {
    const float* x   = (const float*)d_in[0];
    const float* W_i = (const float*)d_in[1];
    const float* b_i = (const float*)d_in[2];
    const float* W_f = (const float*)d_in[3];
    const float* b_f = (const float*)d_in[4];
    const float* W_o = (const float*)d_in[5];
    const float* b_o = (const float*)d_in[6];
    const float* W_u = (const float*)d_in[7];
    const float* b_u = (const float*)d_in[8];

    // Workspace layout: [64KB packed f16 weights][h0][c0][h1][c1] (fp32)
    _Float16* wpack = (_Float16*)d_ws;
    float* fbase = (float*)((char*)d_ws + 65536);
    const size_t LM = (size_t)BATCH * 2048 * HDIM;   // max level elems (d=11)
    float* hb[2] = { fbase,          fbase + 2 * LM };
    float* cb[2] = { fbase + LM,     fbase + 3 * LM };

    pack_weights_kernel<<<128, 256, 0, stream>>>(W_i, W_f, W_o, W_u, wpack);

    int wr = 0;
    for (int d = DEPTH - 1; d >= 0; --d) {
        const int n = 1 << d;
        const int blocks = 4 * n;                    // (256*n/16 tiles) / 4 waves
        const int rd = wr ^ 1;
        float* root_out = (d == 0) ? (float*)d_out : nullptr;
        if (d == DEPTH - 1) {
            tree_level_kernel<true><<<blocks, 128, 0, stream>>>(
                x, wpack, b_i, b_f, b_o, b_u,
                nullptr, nullptr, hb[wr], cb[wr], n, d, root_out);
        } else {
            tree_level_kernel<false><<<blocks, 128, 0, stream>>>(
                x, wpack, b_i, b_f, b_o, b_u,
                hb[rd], cb[rd], hb[wr], cb[wr], n, d, root_out);
        }
        wr ^= 1;
    }
}
